// InterleavedHybridPRINet_16286515986770
// MI455X (gfx1250) — matrix operand order repaired and checked by reading, not compiled, not run on hardware
//
#include <hip/hip_runtime.h>
#include <math.h>

typedef __attribute__((ext_vector_type(16))) _Float16 v16h;
typedef __attribute__((ext_vector_type(4)))  _Float16 v4h;
typedef __attribute__((ext_vector_type(2)))  _Float16 v2h;
typedef __attribute__((ext_vector_type(2)))  __fp16   v2hf;
typedef __attribute__((ext_vector_type(8)))  float    v8f;

#define TWO_PI_F 6.283185307179586f

static __device__ __forceinline__ v8f wmma_f16(v16h a, v16h b, v8f c) {
  return __builtin_amdgcn_wmma_f32_16x16x32_f16(false, a, false, b, (short)0, c, false, false);
}
static __device__ __forceinline__ v2h pk2(float x, float y) {
  union { v2hf a; v2h b; } u;
  u.a = __builtin_amdgcn_cvt_pkrtz(x, y);   // v_cvt_pk_rtz_f16_f32
  return u.b;
}
static __device__ __forceinline__ v8f vzero8() { v8f z = {}; return z; }

// 16 contiguous f16 -> v16h
static __device__ __forceinline__ v16h load16(const _Float16* p) {
  union { v16h v; v4h q[4]; } u;
  u.q[0] = *(const v4h*)(p);
  u.q[1] = *(const v4h*)(p + 4);
  u.q[2] = *(const v4h*)(p + 8);
  u.q[3] = *(const v4h*)(p + 12);
  return u.v;
}
// A-fragment: [kof..kof+15] from a row
static __device__ __forceinline__ v16h load_a16(const _Float16* row, int kof) {
  union { v16h v; v4h q[4]; } u;
  u.q[0] = *(const v4h*)(row + kof);
  u.q[1] = *(const v4h*)(row + kof + 4);
  u.q[2] = *(const v4h*)(row + kof + 8);
  u.q[3] = *(const v4h*)(row + kof + 12);
  return u.v;
}

// ---------------------------------------------------------------------------
// Tiled WMMA GEMM, software pipelined: regs <- global(tile k+1) while
// WMMAs consume LDS(tile k).  block 256 thr (8 wave32), block tile 128x128,
// wave tile 32x64, BK=32.  B tile staged TRANSPOSED (Bst[n][k]).
// act: 0=none, 1=exact GELU, 2=ReLU.  K % 32 == 0 always.
// ---------------------------------------------------------------------------
#define BM 128
#define BN 128
#define BK 32
#define AKP 36
#define BKP 36

template <bool ALIGNED>
__global__ __launch_bounds__(256) void gemm_kernel(
    const float* __restrict__ A, const float* __restrict__ B,
    const float* __restrict__ bias, const float* __restrict__ resid,
    float* __restrict__ C, int M, int N, int K, int act)
{
  __shared__ _Float16 As[BM][AKP];    // 9216 B
  __shared__ _Float16 Bst[BN][BKP];   // 9216 B  (Bst[n][k])

  const int tid  = threadIdx.x;
  const int lane = tid & 31;
  const int wave = tid >> 5;
  const int wm   = wave >> 1;          // 0..3
  const int wn   = wave & 1;           // 0..1
  const int row0 = blockIdx.y * BM;
  const int col0 = blockIdx.x * BN;
  const int fm   = lane & 15;
  const int kof  = (lane >> 4) * 16;
  const int kbsel= lane & 16;

  v8f acc[2][4];
#pragma unroll
  for (int i = 0; i < 2; ++i)
#pragma unroll
    for (int j = 0; j < 4; ++j) acc[i][j] = vzero8();

  // register staging for the software pipeline
  float4 ra[4];          // A: 4 vec4 per thread
  float4 rb0[2], rb1[2]; // B: 2 k-row-pairs x vec4 per thread

  auto load_regs = [&](int kb) {
#pragma unroll
    for (int it = 0; it < 4; ++it) {
      int vi = tid + it * 256;
      int m  = vi >> 3;
      int k4 = (vi & 7) << 2;
      if (ALIGNED) {
        ra[it] = *(const float4*)(A + (size_t)(row0 + m) * K + kb + k4);
      } else {
        int gm = row0 + m;
        bool ok = gm < M;
        if (gm > M - 1) gm = M - 1;
        float4 r = *(const float4*)(A + (size_t)gm * K + kb + k4);
        if (!ok) { r.x = 0.f; r.y = 0.f; r.z = 0.f; r.w = 0.f; }
        ra[it] = r;
      }
    }
#pragma unroll
    for (int it = 0; it < 2; ++it) {
      int vi = tid + it * 256;           // 0..511
      int k2 = (vi >> 5) << 1;           // even k
      int n4 = (vi & 31) << 2;
      int gn0 = col0 + n4;
      if (ALIGNED) {
        rb0[it] = *(const float4*)(B + (size_t)(kb + k2) * N + gn0);
        rb1[it] = *(const float4*)(B + (size_t)(kb + k2 + 1) * N + gn0);
      } else if (gn0 + 4 <= N) {
        rb0[it] = *(const float4*)(B + (size_t)(kb + k2) * N + gn0);
        rb1[it] = *(const float4*)(B + (size_t)(kb + k2 + 1) * N + gn0);
      } else {
        float t0[4], t1[4];
#pragma unroll
        for (int i = 0; i < 4; ++i) {
          int gn = gn0 + i;
          bool ok = gn < N;
          t0[i] = ok ? B[(size_t)(kb + k2) * N + gn] : 0.f;
          t1[i] = ok ? B[(size_t)(kb + k2 + 1) * N + gn] : 0.f;
        }
        rb0[it] = make_float4(t0[0], t0[1], t0[2], t0[3]);
        rb1[it] = make_float4(t1[0], t1[1], t1[2], t1[3]);
      }
    }
  };

  auto store_lds = [&]() {
#pragma unroll
    for (int it = 0; it < 4; ++it) {
      int vi = tid + it * 256;
      int m  = vi >> 3;
      int k4 = (vi & 7) << 2;
      float4 r = ra[it];
      v2h p0 = pk2(r.x, r.y), p1 = pk2(r.z, r.w);
      v4h q; q[0] = p0[0]; q[1] = p0[1]; q[2] = p1[0]; q[3] = p1[1];
      *(v4h*)&As[m][k4] = q;
    }
#pragma unroll
    for (int it = 0; it < 2; ++it) {
      int vi = tid + it * 256;
      int k2 = (vi >> 5) << 1;
      int n4 = (vi & 31) << 2;
      float4 r0 = rb0[it], r1 = rb1[it];
      *(v2h*)&Bst[n4 + 0][k2] = pk2(r0.x, r1.x);
      *(v2h*)&Bst[n4 + 1][k2] = pk2(r0.y, r1.y);
      *(v2h*)&Bst[n4 + 2][k2] = pk2(r0.z, r1.z);
      *(v2h*)&Bst[n4 + 3][k2] = pk2(r0.w, r1.w);
    }
  };

  load_regs(0);
  for (int kb = 0; kb < K; kb += BK) {
    store_lds();
    __syncthreads();
    if (kb + BK < K) load_regs(kb + BK);   // global loads in flight during WMMAs

    v16h a0 = load_a16(&As[wm * 32 + fm][0], kof);
    v16h a1 = load_a16(&As[wm * 32 + 16 + fm][0], kof);
#pragma unroll
    for (int j = 0; j < 4; ++j) {
      v16h bf = load16(&Bst[wn * 64 + j * 16 + fm][kbsel]);
      acc[0][j] = wmma_f16(a0, bf, acc[0][j]);
      acc[1][j] = wmma_f16(a1, bf, acc[1][j]);
    }
    __syncthreads();
  }

  // ---- epilogue ----
  const int rb = (lane & 16) ? 8 : 0;
#pragma unroll
  for (int i = 0; i < 2; ++i)
#pragma unroll
    for (int j = 0; j < 4; ++j) {
      int gn = col0 + wn * 64 + j * 16 + fm;
#pragma unroll
      for (int r = 0; r < 8; ++r) {
        int gm = row0 + wm * 32 + i * 16 + r + rb;
        if (ALIGNED || (gm < M && gn < N)) {
          float v = acc[i][j][r];
          if (bias) v += bias[gn];
          if (act == 1)      v = 0.5f * v * (1.0f + erff(v * 0.70710678118654752f));
          else if (act == 2) v = fmaxf(v, 0.0f);
          if (resid) v += resid[(size_t)gm * N + gn];
          C[(size_t)gm * N + gn] = v;
        }
      }
    }
}

// ---------------------------------------------------------------------------
// LayerNorm over D=512, one wave32 per row.
// ---------------------------------------------------------------------------
__global__ __launch_bounds__(256) void ln_kernel(
    const float* __restrict__ X, float* __restrict__ Y,
    const float* __restrict__ s, const float* __restrict__ b, int M)
{
  int row  = blockIdx.x * 8 + (threadIdx.x >> 5);
  int lane = threadIdx.x & 31;
  if (row >= M) return;
  const float* x = X + (size_t)row * 512;
  float v[16];
  float sum = 0.f;
#pragma unroll
  for (int t = 0; t < 16; t += 4) {
    float4 r = *(const float4*)(x + lane * 16 + t);
    v[t] = r.x; v[t + 1] = r.y; v[t + 2] = r.z; v[t + 3] = r.w;
    sum += r.x + r.y + r.z + r.w;
  }
#pragma unroll
  for (int m = 1; m < 32; m <<= 1) sum += __shfl_xor(sum, m, 32);
  float mean = sum * (1.0f / 512.0f);
  float var = 0.f;
#pragma unroll
  for (int t = 0; t < 16; ++t) { float d = v[t] - mean; var += d * d; }
#pragma unroll
  for (int m = 1; m < 32; m <<= 1) var += __shfl_xor(var, m, 32);
  float inv = rsqrtf(var * (1.0f / 512.0f) + 1e-5f);
  float* y = Y + (size_t)row * 512;
#pragma unroll
  for (int t = 0; t < 16; ++t) {
    int d = lane * 16 + t;
    y[d] = (v[t] - mean) * inv * s[d] + b[d];
  }
}

// ---------------------------------------------------------------------------
// Oscillator dynamics for all 6 layers (tiny; exact fmodf kept).
// ---------------------------------------------------------------------------
__global__ void phase_kernel(const float* __restrict__ Ph,
                             const float* __restrict__ omega,
                             const float* __restrict__ amp_rate,
                             float* __restrict__ token)
{
  int b = blockIdx.x;
  int osc = threadIdx.x;
  if (osc >= 44) return;
  float ph = 0.f;
#pragma unroll
  for (int hh = 0; hh < 8; ++hh) {
    float m = fmodf(Ph[(size_t)b * 2048 + osc * 8 + hh], TWO_PI_F);
    if (m < 0.f) m += TWO_PI_F;
    ph += m;
  }
  ph *= 0.125f;
  float amp = 1.f;
  float w = omega[osc], ar = amp_rate[osc];
  for (int l = 0; l < 6; ++l) {
#pragma unroll
    for (int s = 0; s < 3; ++s) {
      ph = fmodf(ph + 0.01f * TWO_PI_F * w * amp, TWO_PI_F);
      if (ph < 0.f) ph += TWO_PI_F;
      amp += 0.01f * ar * (1.f - amp);
    }
    for (int t = osc; t < 256; t += 44)
      token[((size_t)l * 64 + b) * 256 + t] = ph;
  }
}

// ---------------------------------------------------------------------------
// Fused attention per (b, h, row-half). 8 waves; each wave owns 16 Q rows.
// K staged [t][dh] f16; V staged TRANSPOSED [dh][t] f16; scores strip in LDS.
// Fast HW transcendentals (v_cos_f32 / v_exp_f32) for bias + softmax.
// LDS = 32K(k) + 33K(vT) + 1K(ph) + 0.5K(rowsum) + 128K(scores) = 199168 B.
// ---------------------------------------------------------------------------
#define VTP 264
#define ATTN_LDS (32768 + 33792 + 1024 + 512 + 131072)

__global__ __launch_bounds__(256) void attn_kernel(
    const float* __restrict__ Q, const float* __restrict__ K,
    const float* __restrict__ V, const float* __restrict__ PP,
    const float* __restrict__ token, const float* __restrict__ alpha,
    int layer, float* __restrict__ O)
{
  extern __shared__ char smem[];
  _Float16* kbuf  = (_Float16*)smem;                        // [256][64]
  _Float16* vbufT = (_Float16*)(smem + 32768);              // [64][VTP]
  float* ph       = (float*)(smem + 32768 + 33792);         // [256]
  float* rowsum   = (float*)(smem + 32768 + 33792 + 1024);  // [128]
  float* sc       = (float*)(smem + 32768 + 33792 + 1536);  // [8][16][256]

  const int tid = threadIdx.x, lane = tid & 31, wave = tid >> 5;
  const int bh = blockIdx.x >> 1, qhalf = blockIdx.x & 1;
  const int b = bh >> 3, hh = bh & 7;
  const float alph = alpha[layer];
  const size_t base = ((size_t)b * 256) * 512 + hh * 64;

  // ---- stage K (row-major) and V (transposed), f32 -> f16, vectorized ----
  for (int vi = tid; vi < 256 * 16; vi += 256) {
    int t = vi >> 4, d4 = (vi & 15) << 2;
    float4 r = *(const float4*)(K + base + (size_t)t * 512 + d4);
    v2h p0 = pk2(r.x, r.y), p1 = pk2(r.z, r.w);
    v4h q; q[0] = p0[0]; q[1] = p0[1]; q[2] = p1[0]; q[3] = p1[1];
    *(v4h*)&kbuf[t * 64 + d4] = q;
    float4 s = *(const float4*)(V + base + (size_t)t * 512 + d4);
    vbufT[(d4 + 0) * VTP + t] = (_Float16)s.x;
    vbufT[(d4 + 1) * VTP + t] = (_Float16)s.y;
    vbufT[(d4 + 2) * VTP + t] = (_Float16)s.z;
    vbufT[(d4 + 3) * VTP + t] = (_Float16)s.w;
  }
  if (tid < 256)
    ph[tid] = token[((size_t)layer * 64 + b) * 256 + tid] +
              PP[((size_t)b * 256 + tid) * 8 + hh];
  __syncthreads();

  const int qrow0 = qhalf * 128 + wave * 16;
  const int fm  = lane & 15;
  const int kof = (lane >> 4) * 16;
  const int kbsel = lane & 16;
  const int rb = (lane & 16) ? 8 : 0;

  // ---- Q fragments (two K-chunks over DH=64) ----
  v16h a0, a1;
  {
    const float* qp = Q + base + (size_t)(qrow0 + fm) * 512;
    union { v16h v; v2h h[8]; } u0, u1;
    float4 x0 = *(const float4*)(qp + kof);
    float4 x1 = *(const float4*)(qp + kof + 4);
    float4 x2 = *(const float4*)(qp + kof + 8);
    float4 x3 = *(const float4*)(qp + kof + 12);
    u0.h[0] = pk2(x0.x, x0.y); u0.h[1] = pk2(x0.z, x0.w);
    u0.h[2] = pk2(x1.x, x1.y); u0.h[3] = pk2(x1.z, x1.w);
    u0.h[4] = pk2(x2.x, x2.y); u0.h[5] = pk2(x2.z, x2.w);
    u0.h[6] = pk2(x3.x, x3.y); u0.h[7] = pk2(x3.z, x3.w);
    float4 y0 = *(const float4*)(qp + 32 + kof);
    float4 y1 = *(const float4*)(qp + 32 + kof + 4);
    float4 y2 = *(const float4*)(qp + 32 + kof + 8);
    float4 y3 = *(const float4*)(qp + 32 + kof + 12);
    u1.h[0] = pk2(y0.x, y0.y); u1.h[1] = pk2(y0.z, y0.w);
    u1.h[2] = pk2(y1.x, y1.y); u1.h[3] = pk2(y1.z, y1.w);
    u1.h[4] = pk2(y2.x, y2.y); u1.h[5] = pk2(y2.z, y2.w);
    u1.h[6] = pk2(y3.x, y3.y); u1.h[7] = pk2(y3.z, y3.w);
    a0 = u0.v; a1 = u1.v;
  }

  float* myc = sc + wave * 16 * 256;
  float phq[8];
#pragma unroll
  for (int r = 0; r < 8; ++r) phq[r] = ph[qrow0 + r + rb];

  // ---- pass 1: scores strip (16 x 256) + phase bias (v_cos_f32) ----
  for (int j = 0; j < 16; ++j) {
    const int col = j * 16 + fm;
    const _Float16* kp = kbuf + col * 64 + kbsel;
    v16h b0 = load16(kp);
    v16h b1 = load16(kp + 32);
    v8f c = vzero8();
    c = wmma_f16(a0, b0, c);
    c = wmma_f16(a1, b1, c);
    float pc = ph[col];
#pragma unroll
    for (int r = 0; r < 8; ++r)
      myc[(r + rb) * 256 + col] = c[r] * 0.125f + alph * __cosf(phq[r] - pc);
  }
  __syncthreads();

  // ---- pass 2: softmax (lane pair per row, float4 scans, v_exp_f32) ----
  {
    const int row = lane & 15;
    const int seg = (lane >> 4) * 128;
    float* rp = myc + row * 256 + seg;
    float mx = -1e30f;
    for (int i = 0; i < 128; i += 4) {
      float4 r = *(const float4*)(rp + i);
      mx = fmaxf(mx, fmaxf(fmaxf(r.x, r.y), fmaxf(r.z, r.w)));
    }
    mx = fmaxf(mx, __shfl_xor(mx, 16, 32));
    float s = 0.f;
    for (int i = 0; i < 128; i += 4) {
      float4 r = *(const float4*)(rp + i);
      r.x = __expf(r.x - mx); r.y = __expf(r.y - mx);
      r.z = __expf(r.z - mx); r.w = __expf(r.w - mx);
      s += r.x + r.y + r.z + r.w;
      *(float4*)(rp + i) = r;
    }
    s += __shfl_xor(s, 16, 32);
    if (lane < 16) rowsum[wave * 16 + row] = s;
  }
  __syncthreads();

  // ---- pass 3: O = P @ V (K=256, 4 N-tiles over DH=64) ----
  v8f oc0 = vzero8(), oc1 = vzero8(), oc2 = vzero8(), oc3 = vzero8();
  for (int kc = 0; kc < 256; kc += 32) {
    const float* pr = myc + fm * 256 + kc;
    union { v16h v; v2h h[8]; } ua;
    float4 x0 = *(const float4*)(pr + kof);
    float4 x1 = *(const float4*)(pr + kof + 4);
    float4 x2 = *(const float4*)(pr + kof + 8);
    float4 x3 = *(const float4*)(pr + kof + 12);
    ua.h[0] = pk2(x0.x, x0.y); ua.h[1] = pk2(x0.z, x0.w);
    ua.h[2] = pk2(x1.x, x1.y); ua.h[3] = pk2(x1.z, x1.w);
    ua.h[4] = pk2(x2.x, x2.y); ua.h[5] = pk2(x2.z, x2.w);
    ua.h[6] = pk2(x3.x, x3.y); ua.h[7] = pk2(x3.z, x3.w);
    v16h a = ua.v;
    const _Float16* vp = vbufT + (size_t)fm * VTP + kc + kbsel;
    oc0 = wmma_f16(a, load16(vp + 0 * 16 * VTP), oc0);
    oc1 = wmma_f16(a, load16(vp + 1 * 16 * VTP), oc1);
    oc2 = wmma_f16(a, load16(vp + 2 * 16 * VTP), oc2);
    oc3 = wmma_f16(a, load16(vp + 3 * 16 * VTP), oc3);
  }

  v8f ocs[4] = {oc0, oc1, oc2, oc3};
#pragma unroll
  for (int r = 0; r < 8; ++r) {
    int qr = qrow0 + r + rb;
    float inv = 1.f / rowsum[wave * 16 + r + rb];
#pragma unroll
    for (int n = 0; n < 4; ++n)
      O[base + (size_t)qr * 512 + n * 16 + fm] = ocs[n][r] * inv;
  }
}

// ---------------------------------------------------------------------------
// Mean-pool over T then LayerNorm; one block per batch row.
// ---------------------------------------------------------------------------
__global__ __launch_bounds__(256) void pool_ln_kernel(
    const float* __restrict__ h, const float* __restrict__ pn_s,
    const float* __restrict__ pn_b, float* __restrict__ outp)
{
  __shared__ float pooled[512];
  __shared__ float red[256];
  int b = blockIdx.x, tid = threadIdx.x;
  float s0 = 0.f, s1 = 0.f;
  for (int t = 0; t < 256; ++t) {
    const float* hp = h + ((size_t)b * 256 + t) * 512;
    s0 += hp[tid]; s1 += hp[tid + 256];
  }
  pooled[tid] = s0 * (1.f / 256.f);
  pooled[tid + 256] = s1 * (1.f / 256.f);
  __syncthreads();
  red[tid] = pooled[tid] + pooled[tid + 256]; __syncthreads();
  for (int off = 128; off > 0; off >>= 1) { if (tid < off) red[tid] += red[tid + off]; __syncthreads(); }
  float mean = red[0] * (1.f / 512.f); __syncthreads();
  float d0 = pooled[tid] - mean, d1 = pooled[tid + 256] - mean;
  red[tid] = d0 * d0 + d1 * d1; __syncthreads();
  for (int off = 128; off > 0; off >>= 1) { if (tid < off) red[tid] += red[tid + off]; __syncthreads(); }
  float inv = rsqrtf(red[0] * (1.f / 512.f) + 1e-5f);
  outp[(size_t)b * 512 + tid]       = d0 * inv * pn_s[tid] + pn_b[tid];
  outp[(size_t)b * 512 + tid + 256] = d1 * inv * pn_s[tid + 256] + pn_b[tid + 256];
}

// ---------------------------------------------------------------------------
// clip(+-50) + log_softmax over 1000; one block per batch row.
// ---------------------------------------------------------------------------
__global__ __launch_bounds__(256) void logsoftmax_kernel(
    const float* __restrict__ logits, float* __restrict__ out)
{
  __shared__ float red[256];
  int b = blockIdx.x, tid = threadIdx.x;
  const float* lp = logits + (size_t)b * 1000;
  float mx = -1e30f;
  for (int c = tid; c < 1000; c += 256) {
    float v = fminf(fmaxf(lp[c], -50.f), 50.f);
    mx = fmaxf(mx, v);
  }
  red[tid] = mx; __syncthreads();
  for (int off = 128; off > 0; off >>= 1) { if (tid < off) red[tid] = fmaxf(red[tid], red[tid + off]); __syncthreads(); }
  mx = red[0]; __syncthreads();
  float s = 0.f;
  for (int c = tid; c < 1000; c += 256) {
    float v = fminf(fmaxf(lp[c], -50.f), 50.f);
    s += __expf(v - mx);
  }
  red[tid] = s; __syncthreads();
  for (int off = 128; off > 0; off >>= 1) { if (tid < off) red[tid] += red[tid + off]; __syncthreads(); }
  float lse = __logf(red[0]) + mx;
  for (int c = tid; c < 1000; c += 256) {
    float v = fminf(fmaxf(lp[c], -50.f), 50.f);
    out[(size_t)b * 1000 + c] = v - lse;
  }
}

// ---------------------------------------------------------------------------
extern "C" void kernel_launch(void* const* d_in, const int* in_sizes, int n_in,
                              void* d_out, int out_size, void* d_ws, size_t ws_size,
                              hipStream_t stream)
{
  (void)in_sizes; (void)n_in; (void)out_size; (void)ws_size;
  const float* x        = (const float*)d_in[0];
  const float* W_in     = (const float*)d_in[1];
  const float* b_in     = (const float*)d_in[2];
  const float* W_ph     = (const float*)d_in[3];
  const float* b_ph     = (const float*)d_in[4];
  const float* omega    = (const float*)d_in[5];
  const float* amp_rate = (const float*)d_in[6];
  const float* alpha    = (const float*)d_in[7];
  const float* Wq  = (const float*)d_in[8];
  const float* Wk  = (const float*)d_in[9];
  const float* Wv  = (const float*)d_in[10];
  const float* Wo  = (const float*)d_in[11];
  const float* Wpp = (const float*)d_in[12];
  const float* Wf1 = (const float*)d_in[13];
  const float* Wf2 = (const float*)d_in[14];
  const float* bq  = (const float*)d_in[15];
  const float* bk  = (const float*)d_in[16];
  const float* bv  = (const float*)d_in[17];
  const float* bo  = (const float*)d_in[18];
  const float* bpp = (const float*)d_in[19];
  const float* bf1 = (const float*)d_in[20];
  const float* bf2 = (const float*)d_in[21];
  const float* ln1_b = (const float*)d_in[22];
  const float* ln2_b = (const float*)d_in[23];
  const float* ln1_s = (const float*)d_in[24];
  const float* ln2_s = (const float*)d_in[25];
  const float* pn_s  = (const float*)d_in[26];
  const float* pn_b  = (const float*)d_in[27];
  const float* Wc1 = (const float*)d_in[28];
  const float* bc1 = (const float*)d_in[29];
  const float* Wc2 = (const float*)d_in[30];
  const float* bc2 = (const float*)d_in[31];

  const int Bd = 64, T = 256, D = 512, H = 8, L = 6, NC = 1000;
  const size_t MT = (size_t)Bd * T;   // 16384 rows

  char* wp = (char*)d_ws;
  auto alloc = [&](size_t bytes) -> float* {
    float* p = (float*)wp;
    wp += (bytes + 255) & ~(size_t)255;
    return p;
  };
  float* h    = alloc(MT * D * 4);
  float* hn   = alloc(MT * D * 4);
  float* obuf = alloc(MT * D * 4);
  float* big  = alloc(MT * 4 * D * 4);      // QKV during attention, FFN hidden after
  float* q = big;
  float* k = big + MT * D;
  float* v = big + 2 * MT * D;
  float* Ph     = alloc((size_t)Bd * T * H * 4);
  float* PP     = alloc(MT * H * 4);
  float* token  = alloc((size_t)L * Bd * T * 4);
  float* pooledN= alloc((size_t)Bd * D * 4);
  float* c1     = alloc((size_t)Bd * D * 4);
  float* logits = alloc((size_t)Bd * NC * 4);

  dim3 blk(256);
  auto gemm = [&](const float* A, const float* Bm, const float* bias,
                  const float* resid, float* C, int M, int N, int K, int act) {
    dim3 grid((N + BN - 1) / BN, (M + BM - 1) / BM);
    if ((M % BM == 0) && (N % BN == 0))
      gemm_kernel<true><<<grid, blk, 0, stream>>>(A, Bm, bias, resid, C, M, N, K, act);
    else
      gemm_kernel<false><<<grid, blk, 0, stream>>>(A, Bm, bias, resid, C, M, N, K, act);
  };

  // input projections
  gemm(x, W_in, b_in, nullptr, h,  Bd, T * D, 1024, 0);
  gemm(x, W_ph, b_ph, nullptr, Ph, Bd, T * H, 1024, 0);
  phase_kernel<<<dim3(Bd), dim3(64), 0, stream>>>(Ph, omega, amp_rate, token);

  for (int l = 0; l < L; ++l) {
    ln_kernel<<<dim3((int)(MT / 8)), blk, 0, stream>>>(h, hn, ln1_s + l * D, ln1_b + l * D, (int)MT);
    gemm(hn, Wq  + (size_t)l * D * D, bq  + l * D, nullptr, q,  (int)MT, D, D, 0);
    gemm(hn, Wk  + (size_t)l * D * D, bk  + l * D, nullptr, k,  (int)MT, D, D, 0);
    gemm(hn, Wv  + (size_t)l * D * D, bv  + l * D, nullptr, v,  (int)MT, D, D, 0);
    gemm(hn, Wpp + (size_t)l * D * H, bpp + l * H, nullptr, PP, (int)MT, H, D, 0);
    attn_kernel<<<dim3(Bd * H * 2), blk, ATTN_LDS, stream>>>(q, k, v, PP, token, alpha, l, obuf);
    gemm(obuf, Wo + (size_t)l * D * D, bo + l * D, h, h, (int)MT, D, D, 0);
    ln_kernel<<<dim3((int)(MT / 8)), blk, 0, stream>>>(h, hn, ln2_s + l * D, ln2_b + l * D, (int)MT);
    gemm(hn,  Wf1 + (size_t)l * D * 4 * D, bf1 + (size_t)l * 4 * D, nullptr, big, (int)MT, 4 * D, D, 1);
    gemm(big, Wf2 + (size_t)l * 4 * D * D, bf2 + l * D,             h,       h,   (int)MT, D, 4 * D, 0);
  }

  pool_ln_kernel<<<dim3(Bd), blk, 0, stream>>>(h, pn_s, pn_b, pooledN);
  gemm(pooledN, Wc1, bc1, nullptr, c1,     Bd, D,  D, 2);
  gemm(c1,      Wc2, bc2, nullptr, logits, Bd, NC, D, 0);
  logsoftmax_kernel<<<dim3(Bd), blk, 0, stream>>>(logits, (float*)d_out);
}
